// xLSTMEncoderV3_45921790328945
// MI455X (gfx1250) — matrix operand order repaired and checked by reading, not compiled, not run on hardware
//
#include <hip/hip_runtime.h>

// xLSTM scan, MI455X (gfx1250):
//  - 32 persistent workgroups, one per batch element (scan is batch-separable).
//  - C_m (256x256 f32) lives in LDS (stride-260 rows: 16B aligned, bank-skewed).
//  - Weights pre-packed once to bf16 in WMMA B-fragment layout in d_ws.
//  - All dense projections via v_wmma_f32_16x16x32_bf16 (single live A-row).
//  - WGP-scope global_prefetch_b8 (inline asm, scope 0 = pull into WGP$) runs
//    2 k-steps ahead of the WMMA stream: 32 KB/WGP in flight, fits WGP$, and
//    the B-fragment loads hit near cache instead of eating L2 latency.
//  - q / Wo / C_m.q readout only at the final timestep (matches reference).

#define B_    32
#define T_    256
#define H_    256
#define CSTR  260   // C_m row stride in floats: 16B aligned, skewed banks

// packed-weight offsets in d_ws (bf16 elements)
#define OFF_WS   0          // W_s : K=256, N=1024 -> 262144
#define OFF_RS   262144     // R_s : K=256, N=1024 -> 262144
#define OFF_QKVO 524288     // m*65536 : m0=Wk, m1=Wv, m2=Wq, m3=Wo

#define TILE_EL  512        // bf16 elements per 32x16 B tile
#define NT_EL    (8 * TILE_EL)          // elements per N-tile K-stream
#define PAIR_EL  (16 * NT_EL)           // stream advance to next tile-pair

typedef __attribute__((ext_vector_type(16))) __bf16 v16bf;
typedef __attribute__((ext_vector_type(8)))  float  v8f;
typedef __attribute__((ext_vector_type(4)))  float  v4f;

__device__ __forceinline__ unsigned short f2bf(float f) {
  unsigned int u = __float_as_uint(f);
  u += 0x7FFFu + ((u >> 16) & 1u);          // round-to-nearest-even
  return (unsigned short)(u >> 16);
}
__device__ __forceinline__ __bf16 us2bf(unsigned short u) {
  return __builtin_bit_cast(__bf16, u);
}
__device__ __forceinline__ float sigm(float x) { return 1.0f / (1.0f + __expf(-x)); }

// WGP-scope prefetch (scope 0: fill all cache levels incl. WGP$).
// __builtin_prefetch only emits SCOPE_SE (GL2-only), so use inline asm.
__device__ __forceinline__ void pf(const unsigned short* p) {
  asm volatile("global_prefetch_b8 %0, off" :: "v"((const void*)p) : "memory");
}

// block-wide sum over 256 threads (8 wave32s)
__device__ __forceinline__ float rsum(float v, float* scr) {
  #pragma unroll
  for (int o = 16; o > 0; o >>= 1) v += __shfl_xor(v, o, 32);
  const int w = threadIdx.x >> 5;
  __syncthreads();                           // protect scr from previous call
  if ((threadIdx.x & 31) == 0) scr[w] = v;
  __syncthreads();
  float s = 0.f;
  #pragma unroll
  for (int k = 0; k < 8; ++k) s += scr[k];
  return s;
}

// A fragment: 16x32 bf16, single live row M=0 holding vec[kt*32 .. kt*32+31].
// ISA 16-bit A layout: lane l holds row M=l%16; halves 0..7 -> K=(l>>4)*8+0..7,
// halves 8..15 -> K=(l>>4)*8+16..23.  Only lanes with l%16==0 are row 0.
__device__ __forceinline__ v16bf build_a(const unsigned short* sh, int kt, int lane) {
  const int act = ((lane & 15) == 0) ? 0xFFFF : 0;
  const int kb  = kt * 32 + ((lane >> 4) << 3);
  v16bf a;
  #pragma unroll
  for (int h = 0; h < 8; ++h) {
    a[h]     = us2bf((unsigned short)(sh[kb + h]      & act));
    a[h + 8] = us2bf((unsigned short)(sh[kb + 16 + h] & act));
  }
  return a;
}

__device__ __forceinline__ v16bf loadB(const unsigned short* p) {
  return *reinterpret_cast<const v16bf*>(p);   // 32B contiguous per lane
}

__device__ __forceinline__ v8f wmma_bf16(v16bf a, v16bf b, v8f c) {
  return __builtin_amdgcn_wmma_f32_16x16x32_bf16(false, a, false, b, (short)0, c,
                                                 false, false);
}

// ---------------------------------------------------------------------------
// Pack f32 row-major [K,N] weight into bf16 WMMA-B-fragment tiles:
// tile (nt,kt) -> 512 bf16 at ((nt*KT+kt)*512); lane l owns 16 contiguous
// halves: element W[kt*32 + (l>>4)*8 + (h&7) + (h>>3)*16][nt*16 + (l&15)].
// ---------------------------------------------------------------------------
__global__ void pack_b_kernel(const float* __restrict__ src,
                              unsigned short* __restrict__ dst, int K, int N) {
  const int e = blockIdx.x * blockDim.x + threadIdx.x;
  if (e >= K * N) return;
  const int KT   = K >> 5;
  const int tile = e >> 9;
  const int w    = e & 511;
  const int lane = w >> 4;
  const int h    = w & 15;
  const int nt   = tile / KT;
  const int kt   = tile - nt * KT;
  const int k    = kt * 32 + ((lane >> 4) << 3) + (h & 7) + ((h >> 3) << 4);
  const int n    = nt * 16 + (lane & 15);
  dst[e] = f2bf(src[k * N + n]);
}

// ---------------------------------------------------------------------------
// Persistent per-batch scan kernel. 256 threads = 8 wave32s.
// ---------------------------------------------------------------------------
__global__ __launch_bounds__(256, 1)
void xlstm_scan_kernel(const float* __restrict__ x,
                       const float* __restrict__ g_s,  const float* __restrict__ bn_s,
                       const float* __restrict__ b_s,
                       const float* __restrict__ g_m,  const float* __restrict__ bn_m,
                       const float* __restrict__ bq,   const float* __restrict__ bk,
                       const float* __restrict__ bv,   const float* __restrict__ bo,
                       const float* __restrict__ wi,   const float* __restrict__ bi,
                       const float* __restrict__ wf,   const float* __restrict__ bff,
                       const float* __restrict__ g_o,  const float* __restrict__ bn_o,
                       const unsigned short* __restrict__ pk,
                       float* __restrict__ out) {
  extern __shared__ char smem[];
  float* Cm       = (float*)smem;                    // 256*260 f32  (~260 KB)
  float* sh_gates = Cm + H_ * CSTR;                  // 1024
  float* sh_q     = sh_gates + 4 * H_;               // 256
  float* sh_k     = sh_q + H_;                       // 256
  float* sh_v     = sh_k + H_;                       // 256
  float* sh_og    = sh_v + H_;                       // 256
  float* scr      = sh_og + H_;                      // 16
  unsigned short* sh_a  = (unsigned short*)(scr + 16);  // 256 bf16 (xn / xm)
  unsigned short* sh_hb = sh_a + H_;                    // 256 bf16 (h_s)

  const int b    = blockIdx.x;
  const int i    = threadIdx.x;       // element / C_m row index
  const int lane = i & 31;
  const int wv   = i >> 5;
  const float invH = 1.0f / (float)H_;
  const float inv_sqrt_h = 0.0625f;   // 1/sqrt(256)

  // per-wave packed-weight stream bases (lane slice folded in)
  const unsigned short* gW = pk + OFF_WS + wv * NT_EL + lane * 16;
  const unsigned short* gR = pk + OFF_RS + wv * NT_EL + lane * 16;
  const unsigned short* gQ = pk + OFF_QKVO + wv * NT_EL + lane * 16;

  // state (per batch element i)
  float hs = 0.f, cs = 0.f, ns = 1.f, nm = 0.f;
  sh_a[i] = 0; sh_hb[i] = 0;
  {
    float* row = Cm + i * CSTR;
    #pragma unroll 4
    for (int j = 0; j < CSTR; ++j) row[j] = 0.f;
  }
  const float bi_v = bi[0], bf_v = bff[0];
  const float gsi = g_s[i], bnsi = bn_s[i];
  const float gmi = g_m[i], bnmi = bn_m[i];
  const float wii = wi[i],  wfi  = wf[i];
  __syncthreads();

  const float* xb = x + (size_t)b * T_ * H_;

  for (int t = 0; t < T_; ++t) {
    const bool last = (t == T_ - 1);
    const float xt = xb[t * H_ + i];

    // warm WGP$ for the first two k-steps of the gates phase (runs during
    // the barrier-heavy LN1 reductions)
    #pragma unroll
    for (int kt = 0; kt < 2; ++kt) {
      pf(gW + kt * TILE_EL);
      pf(gR + kt * TILE_EL);
      pf(gW + 8 * NT_EL + kt * TILE_EL);
      pf(gR + 8 * NT_EL + kt * TILE_EL);
    }

    // ---- LayerNorm 1 (pre-sLSTM) ----
    float mu  = rsum(xt, scr) * invH;
    float d   = xt - mu;
    float var = rsum(d * d, scr) * invH;
    float xn  = d * rsqrtf(var + 1e-5f) * gsi + bnsi;
    sh_a[i] = f2bf(xn);
    __syncthreads();

    // ---- gates = xn@W_s + h_s@R_s + b_s ----
    // 64 N-tiles / 8 waves; two tiles (ntA, ntA+8) per wave -> two
    // independent accumulation chains.  WGP-scope prefetch runs exactly
    // 2 k-steps ahead (32 KB/WGP in flight -> fits WGP$).
    #pragma unroll 1
    for (int p = 0; p < 4; ++p) {
      const int ntA = wv + p * 16;
      const int ntB = ntA + 8;
      const unsigned short* wW0 = pk + OFF_WS + ntA * NT_EL + lane * 16;
      const unsigned short* wR0 = pk + OFF_RS + ntA * NT_EL + lane * 16;
      const unsigned short* wW1 = pk + OFF_WS + ntB * NT_EL + lane * 16;
      const unsigned short* wR1 = pk + OFF_RS + ntB * NT_EL + lane * 16;
      v8f acc0 = {0, 0, 0, 0, 0, 0, 0, 0};
      v8f acc1 = {0, 0, 0, 0, 0, 0, 0, 0};
      #pragma unroll
      for (int kt = 0; kt < 8; ++kt) {
        // prefetch k-step kt+2 (spilling into the next tile-pair's stream);
        // kt is a compile-time constant here (loop fully unrolled)
        if (kt + 2 < 8) {
          const int po = (kt + 2) * TILE_EL;
          pf(wW0 + po); pf(wR0 + po); pf(wW1 + po); pf(wR1 + po);
        } else if (p < 3) {
          const int po = PAIR_EL + (kt + 2 - 8) * TILE_EL;
          pf(wW0 + po); pf(wR0 + po); pf(wW1 + po); pf(wR1 + po);
        }
        const int off = kt * TILE_EL;
        v16bf bW0 = loadB(wW0 + off);
        v16bf bR0 = loadB(wR0 + off);
        v16bf bW1 = loadB(wW1 + off);
        v16bf bR1 = loadB(wR1 + off);
        v16bf aX  = build_a(sh_a,  kt, lane);
        v16bf aH  = build_a(sh_hb, kt, lane);
        acc0 = wmma_bf16(aX, bW0, acc0);
        acc0 = wmma_bf16(aH, bR0, acc0);
        acc1 = wmma_bf16(aX, bW1, acc1);
        acc1 = wmma_bf16(aH, bR1, acc1);
      }
      if (lane < 16) {                 // D row 0 lives in acc[0], lanes 0..15
        const int nA = ntA * 16 + lane;
        const int nB = ntB * 16 + lane;
        sh_gates[nA] = acc0[0] + b_s[nA];
        sh_gates[nB] = acc1[0] + b_s[nB];
      }
    }

    // warm WGP$ for the first two k-steps of the projection phase
    #pragma unroll
    for (int kt = 0; kt < 2; ++kt) {
      pf(gQ + kt * TILE_EL);
      pf(gQ + 8 * NT_EL + kt * TILE_EL);
    }
    __syncthreads();

    // ---- sLSTM pointwise recurrence ----
    const float gi = __expf(sh_gates[i]);
    const float gf = sigm(sh_gates[H_ + i]);
    const float gz = tanhf(sh_gates[2 * H_ + i]);
    const float go = sigm(sh_gates[3 * H_ + i]);
    cs = gf * cs + gi * gz;
    ns = gf * ns + gi;
    hs = go * cs / ns;
    const float h1 = xt + hs;
    sh_hb[i] = f2bf(hs);               // consumed by next step's R_s pass

    // ---- LayerNorm 2 (pre-mLSTM) ----
    float mu2  = rsum(h1, scr) * invH;
    float d2   = h1 - mu2;
    float var2 = rsum(d2 * d2, scr) * invH;
    float xm   = d2 * rsqrtf(var2 + 1e-5f) * gmi + bnmi;
    sh_a[i] = f2bf(xm);
    __syncthreads();

    // ---- k,v (plus q,o at t=T-1) via WMMA, paired tiles ----
    const int npair = last ? 4 : 2;     // {Wk,Wv[,Wq,Wo]} x 16 tiles, pairs
    #pragma unroll 1
    for (int p = 0; p < npair; ++p) {
      const int t0 = wv + p * 16;
      const int t1 = t0 + 8;
      const int m0 = t0 >> 4, n0t = t0 & 15;
      const int m1 = t1 >> 4, n1t = t1 & 15;
      const unsigned short* wB0 =
          pk + OFF_QKVO + m0 * 65536 + n0t * NT_EL + lane * 16;
      const unsigned short* wB1 =
          pk + OFF_QKVO + m1 * 65536 + n1t * NT_EL + lane * 16;
      const bool more = (p + 1 < npair);   // block-uniform
      v8f acc0 = {0, 0, 0, 0, 0, 0, 0, 0};
      v8f acc1 = {0, 0, 0, 0, 0, 0, 0, 0};
      #pragma unroll
      for (int kt = 0; kt < 8; ++kt) {
        if (kt + 2 < 8) {
          const int po = (kt + 2) * TILE_EL;
          pf(wB0 + po); pf(wB1 + po);
        } else if (more) {
          const int po = PAIR_EL + (kt + 2 - 8) * TILE_EL;
          pf(wB0 + po); pf(wB1 + po);
        }
        const int off = kt * TILE_EL;
        v16bf b0 = loadB(wB0 + off);
        v16bf b1 = loadB(wB1 + off);
        v16bf aX = build_a(sh_a, kt, lane);
        acc0 = wmma_bf16(aX, b0, acc0);
        acc1 = wmma_bf16(aX, b1, acc1);
      }
      if (lane < 16) {
        const int n0 = n0t * 16 + lane;
        const int n1 = n1t * 16 + lane;
        const float r0 = acc0[0];
        const float r1 = acc1[0];
        if      (m0 == 0) sh_k[n0]  = r0 + bk[n0];
        else if (m0 == 1) sh_v[n0]  = r0 + bv[n0];
        else if (m0 == 2) sh_q[n0]  = r0 + bq[n0];
        else              sh_og[n0] = r0 + bo[n0];
        if      (m1 == 0) sh_k[n1]  = r1 + bk[n1];
        else if (m1 == 1) sh_v[n1]  = r1 + bv[n1];
        else if (m1 == 2) sh_q[n1]  = r1 + bq[n1];
        else              sh_og[n1] = r1 + bo[n1];
      }
    }
    __syncthreads();

    // ---- scalar gates + n_m update ----
    const float kk = sh_k[i] * inv_sqrt_h;
    const float di = rsum(xm * wii, scr);
    const float df = rsum(xm * wfi, scr);
    sh_k[i] = kk;                       // store scaled k for the C_m pass
    const float igm = __expf(di + bi_v);
    const float fgm = sigm(df + bf_v);
    nm = fgm * nm + igm * kk;
    __syncthreads();

    // ---- C_m rank-1 update in LDS (thread i owns row i) ----
    float* row = Cm + i * CSTR;
    const float igv = igm * sh_v[i];
    if (!last) {
      #pragma unroll 4
      for (int j = 0; j < H_; j += 4) {
        v4f c4 = *(v4f*)(row + j);
        v4f k4 = *(const v4f*)(sh_k + j);
        c4 = c4 * fgm + k4 * igv;
        *(v4f*)(row + j) = c4;
      }
    } else {
      // final step: fused update + readout h_til = C_m.q, then output LN
      float htil = 0.f;
      #pragma unroll 4
      for (int j = 0; j < H_; j += 4) {
        v4f c4 = *(v4f*)(row + j);
        v4f k4 = *(const v4f*)(sh_k + j);
        v4f q4 = *(const v4f*)(sh_q + j);
        c4 = c4 * fgm + k4 * igv;
        *(v4f*)(row + j) = c4;
        v4f p = c4 * q4;
        htil += p[0] + p[1] + p[2] + p[3];
      }
      const float dq   = rsum(nm * sh_q[i], scr);
      const float den  = fmaxf(fabsf(dq), 1.0f);
      const float hm   = sigm(sh_og[i]) * htil / den;
      const float hout = h1 + hm;
      float mu3  = rsum(hout, scr) * invH;
      float d3   = hout - mu3;
      float var3 = rsum(d3 * d3, scr) * invH;
      out[b * H_ + i] = d3 * rsqrtf(var3 + 1e-5f) * g_o[i] + bn_o[i];
    }
    __syncthreads();
  }
}

// ---------------------------------------------------------------------------
extern "C" void kernel_launch(void* const* d_in, const int* in_sizes, int n_in,
                              void* d_out, int out_size, void* d_ws, size_t ws_size,
                              hipStream_t stream) {
  const float* x    = (const float*)d_in[0];
  const float* g_s  = (const float*)d_in[1];
  const float* bn_s = (const float*)d_in[2];
  const float* W_s  = (const float*)d_in[3];
  const float* R_s  = (const float*)d_in[4];
  const float* b_s  = (const float*)d_in[5];
  const float* g_m  = (const float*)d_in[6];
  const float* bn_m = (const float*)d_in[7];
  const float* Wq   = (const float*)d_in[8];
  const float* bq   = (const float*)d_in[9];
  const float* Wk   = (const float*)d_in[10];
  const float* bk   = (const float*)d_in[11];
  const float* Wv   = (const float*)d_in[12];
  const float* bv   = (const float*)d_in[13];
  const float* wi   = (const float*)d_in[14];
  const float* bi   = (const float*)d_in[15];
  const float* wf   = (const float*)d_in[16];
  const float* bf_  = (const float*)d_in[17];
  const float* Wo   = (const float*)d_in[18];
  const float* bo   = (const float*)d_in[19];
  const float* g_o  = (const float*)d_in[20];
  const float* bn_o = (const float*)d_in[21];
  float* out = (float*)d_out;
  unsigned short* pk = (unsigned short*)d_ws;   // 1.5 MB of bf16 packed weights

  // pack weights (every call; deterministic, inputs never mutated)
  pack_b_kernel<<<1024, 256, 0, stream>>>(W_s, pk + OFF_WS, 256, 1024);
  pack_b_kernel<<<1024, 256, 0, stream>>>(R_s, pk + OFF_RS, 256, 1024);
  pack_b_kernel<<<256, 256, 0, stream>>>(Wk, pk + OFF_QKVO + 0 * 65536, 256, 256);
  pack_b_kernel<<<256, 256, 0, stream>>>(Wv, pk + OFF_QKVO + 1 * 65536, 256, 256);
  pack_b_kernel<<<256, 256, 0, stream>>>(Wq, pk + OFF_QKVO + 2 * 65536, 256, 256);
  pack_b_kernel<<<256, 256, 0, stream>>>(Wo, pk + OFF_QKVO + 3 * 65536, 256, 256);

  // dynamic LDS: C_m (256*260 f32) + staging + bf16 vectors = ~275.5 KB < 320 KB
  const size_t smem = (size_t)(H_ * CSTR + 4 * H_ + 4 * H_ + 16) * sizeof(float)
                    + (size_t)(2 * H_) * sizeof(unsigned short);

  xlstm_scan_kernel<<<dim3(B_), dim3(256), smem, stream>>>(
      x, g_s, bn_s, b_s, g_m, bn_m, bq, bk, bv, bo,
      wi, bi, wf, bf_, g_o, bn_o, pk, out);
}
